// Model_11871289606789
// MI455X (gfx1250) — matrix operand order repaired and checked by reading, not compiled, not run on hardware
//
#include <hip/hip_runtime.h>
#include <hip/hip_bf16.h>

// Problem constants (match reference)
#define Bsz 2048
#define Dd  512
#define Cc  128
#define Tt  151

typedef __attribute__((ext_vector_type(16))) __bf16 v16bf;
typedef __attribute__((ext_vector_type(8)))  float  v8f;
typedef unsigned int   u32;
typedef unsigned short bfraw;
typedef __attribute__((ext_vector_type(4))) u32 u32x4;

union FragU { u32x4 q[2]; v16bf v; };

__device__ __forceinline__ bfraw f2bf(float f) {
  union { float f; u32 u; } x; x.f = f;
  u32 r = x.u + 0x7FFFu + ((x.u >> 16) & 1u);   // round-to-nearest-even
  return (bfraw)(r >> 16);
}

__device__ __forceinline__ float fsigmoid(float x) {
  return 1.f / (1.f + __expf(-x));
}
__device__ __forceinline__ float ftanh(float x) {
  return 2.f / (1.f + __expf(-2.f * x)) - 1.f;
}

// A-matrix fragment (16x32 bf16) from LDS row: lanes 0-15 hold K=k0+[0..15],
// lanes 16-31 hold K=k0+[16..31].
__device__ __forceinline__ v16bf load_fragA(const bfraw* rowbase, int k0, int lane) {
  const int hs = (lane >> 4) << 4;
  const bfraw* p = rowbase + k0 + hs;
  FragU f;
  f.q[0] = *(const u32x4*)(p);
  f.q[1] = *(const u32x4*)(p + 8);
  return f.v;
}

// B-matrix fragment (32x16 bf16): per-lane one contiguous 32B chunk.
__device__ __forceinline__ void loadB(FragU* b, const bfraw* base, u32 elemofs) {
  const bfraw* p = base + elemofs;
  b->q[0] = *(const u32x4*)(p);
  b->q[1] = *(const u32x4*)(p + 8);
}

#define WMMA_BF16(A, B, C) \
  __builtin_amdgcn_wmma_f32_16x16x32_bf16(false, (A), false, (B), (short)0, (C), false, false)

// One GEMM pass: acc[0..3] += h(16xD) x W^T for 4 N-tiles of 16.
// Manually software-pipelined; K-loop unrolling is pinned off so the ping-pong
// buffers stay at 2x4 fragments and nothing spills.
__device__ __forceinline__ void gemm_pass4(v8f acc[4], const bfraw* hrow,
                                           const bfraw* base, const u32 ofs[4],
                                           int lane) {
  FragU b0[4], b1[4];
#pragma unroll
  for (int i = 0; i < 4; ++i) loadB(&b0[i], base, ofs[i]);
#pragma unroll 1
  for (int k0 = 0; k0 < Dd; k0 += 64) {
    v16bf a0 = load_fragA(hrow, k0, lane);
#pragma unroll
    for (int i = 0; i < 4; ++i) loadB(&b1[i], base, ofs[i] + (u32)(k0 + 32));
#pragma unroll
    for (int i = 0; i < 4; ++i) acc[i] = WMMA_BF16(a0, b0[i].v, acc[i]);
    v16bf a1 = load_fragA(hrow, k0 + 32, lane);
#pragma unroll
    for (int i = 0; i < 4; ++i) loadB(&b0[i], base, ofs[i] + (u32)((k0 + 64) & (Dd - 1)));
#pragma unroll
    for (int i = 0; i < 4; ++i) acc[i] = WMMA_BF16(a1, b1[i].v, acc[i]);
  }
}

// ---------------- precompute kernels ----------------

__global__ void cvt_bf16_kernel(const float* __restrict__ src, bfraw* __restrict__ dst, int n) {
  int i = blockIdx.x * blockDim.x + threadIdx.x;
  if (i < n) dst[i] = f2bf(src[i]);
}

// G[c][n] = sum_k embed[c][k]*W_ih[n][k] + b_ih[n] for c<128; G[128][n] = b_ih[n]
__global__ void build_G_kernel(const float* __restrict__ embed, const float* __restrict__ W_ih,
                               const float* __restrict__ b_ih, float* __restrict__ G) {
  int idx = blockIdx.x * blockDim.x + threadIdx.x;
  if (idx >= 129 * (3 * Dd)) return;
  int c = idx / (3 * Dd);
  int n = idx - c * (3 * Dd);
  float s = b_ih[n];
  if (c < Cc) {
    const float* e  = embed + (size_t)c * Dd;
    const float* wi = W_ih + (size_t)n * Dd;
#pragma unroll 4
    for (int k = 0; k < Dd; ++k) s += e[k] * wi[k];
  }
  G[idx] = s;
}

// ---------------- main persistent GRU decoder ----------------
// One workgroup (8 wave32s) owns 16 batch rows for all 151 steps.

__global__ __launch_bounds__(256) void gru_decode_kernel(
    const float* __restrict__ feat,       // [B][D]  (h0)
    const float* __restrict__ b_hh,       // [3D]
    const bfraw* __restrict__ Whh,        // [3D][D] bf16
    const bfraw* __restrict__ Wproj,      // [C][D]  bf16
    const float* __restrict__ G,          // [C+1][3D]  precomputed gi table
    const float* __restrict__ b_proj,     // [C]
    float* __restrict__ logits_out,       // [B][C][T]
    int*   __restrict__ tok_out)          // [B][T]
{
  __shared__ __align__(16) float h_f[16 * Dd];     // 32 KB  f32 hidden state
  __shared__ __align__(16) bfraw h_b[16 * Dd];     // 16 KB  bf16 mirror (WMMA A operand)
  __shared__ __align__(16) float lg[16 * Cc];      //  8 KB  logits tile
  __shared__ __align__(32) float ghr[32 * 32 * 8]; // 32 KB  r-gate staging (per-lane packed)
  __shared__ __align__(32) float ghz[32 * 32 * 8]; // 32 KB  z-gate staging
  __shared__ int tok[16];

  const int tid  = threadIdx.x;
  const int lane = tid & 31;
  const int w    = tid >> 5;          // wave id, 0..7
  const int b0   = blockIdx.x << 4;   // batch tile base row
  const int mrow = lane & 15;         // A-row / B-row / N-col owned by this lane
  const int hs8  = (lane >> 4) << 3;  // C/D: lanes>=16 hold M = 8..15
  const int hsB  = (lane >> 4) << 4;  // B frag: lanes>=16 hold K-half 16..31

  // init: h = feat, tok = 128 (zero start token -> G row 128 = b_ih)
  for (int idx = tid; idx < 16 * Dd; idx += 256) {
    float v = feat[(size_t)(b0 + (idx >> 9)) * Dd + (idx & (Dd - 1))];
    h_f[idx] = v;
    h_b[idx] = f2bf(v);
  }
  if (tid < 16) tok[tid] = Cc;
  __syncthreads();

  const int jbase = w << 2;           // this wave's 4 col-blocks: 32 blocks cover D=512
  const bfraw* hrow = h_b + mrow * Dd;

  // per-block B-fragment element offsets (uniform base per gate added separately)
  u32 ofs[4];
  int gidx[4];                        // per-lane staging slots (8 floats each)
#pragma unroll
  for (int i = 0; i < 4; ++i) {
    ofs[i]  = (u32)((((jbase + i) << 4) + mrow) * Dd + hsB);
    gidx[i] = (((jbase + i) << 5) + lane) << 3;
  }
  const int cproj = (w << 4) + mrow;  // projection column for this lane
  const u32 pofs  = (u32)(cproj * Dd + hsB);
  const float bpc = b_proj[cproj];

  for (int t = 0; t < Tt; ++t) {
    // ---- Phase A1: gh = h @ Whh^T, three pipelined gate passes.
    // r and z results are parked in LDS (same-lane round trip, no barrier
    // needed); only the n-gate accumulators stay in registers.
    {
      v8f ac[4];
#pragma unroll
      for (int i = 0; i < 4; ++i) ac[i] = (v8f){};
      gemm_pass4(ac, hrow, Whh, ofs, lane);
#pragma unroll
      for (int i = 0; i < 4; ++i) *(v8f*)&ghr[gidx[i]] = ac[i];
    }
    {
      v8f ac[4];
#pragma unroll
      for (int i = 0; i < 4; ++i) ac[i] = (v8f){};
      gemm_pass4(ac, hrow, Whh + (size_t)Dd * Dd, ofs, lane);
#pragma unroll
      for (int i = 0; i < 4; ++i) *(v8f*)&ghz[gidx[i]] = ac[i];
    }
    v8f an[4];
#pragma unroll
    for (int i = 0; i < 4; ++i) an[i] = (v8f){};
    gemm_pass4(an, hrow, Whh + (size_t)2 * Dd * Dd, ofs, lane);
    __syncthreads();   // all waves finished reading h_b (old h)

    // ---- Phase A2: gates + hidden update (each lane owns its (m,d) elems) ----
#pragma unroll 1
    for (int i = 0; i < 4; ++i) {
      int d = ((jbase + i) << 4) + mrow;
      float bhr = b_hh[d], bhz = b_hh[Dd + d], bhn = b_hh[2 * Dd + d];
      v8f vr = *(const v8f*)&ghr[gidx[i]];
      v8f vz = *(const v8f*)&ghz[gidx[i]];
#pragma unroll
      for (int e = 0; e < 8; ++e) {
        int m = hs8 + e;
        const float* g = G + (size_t)tok[m] * (3 * Dd);
        float r  = fsigmoid(g[d]      + vr[e]    + bhr);
        float z  = fsigmoid(g[Dd + d] + vz[e]    + bhz);
        float nn = ftanh(g[2 * Dd + d] + r * (an[i][e] + bhn));
        float hn = (1.f - z) * nn + z * h_f[m * Dd + d];
        h_f[m * Dd + d] = hn;
        h_b[m * Dd + d] = f2bf(hn);
      }
    }
    __syncthreads();   // new h_b visible to all waves

    // ---- Phase B: logits = h_new @ Wproj^T (one 16-col tile per wave) ----
    v8f cl = (v8f){};
    {
      FragU p0, p1;
      loadB(&p0, Wproj, pofs);
#pragma unroll 1
      for (int k0 = 0; k0 < Dd; k0 += 64) {
        v16bf a0 = load_fragA(hrow, k0, lane);
        loadB(&p1, Wproj, pofs + (u32)(k0 + 32));
        cl = WMMA_BF16(a0, p0.v, cl);
        v16bf a1 = load_fragA(hrow, k0 + 32, lane);
        loadB(&p0, Wproj, pofs + (u32)((k0 + 64) & (Dd - 1)));
        cl = WMMA_BF16(a1, p1.v, cl);
      }
    }
#pragma unroll
    for (int e = 0; e < 8; ++e) {
      int m = hs8 + e;
      float v = cl[e] + bpc;
      lg[m * Cc + cproj] = v;
      logits_out[((size_t)(b0 + m) * Cc + cproj) * Tt + t] = v;
    }
    __syncthreads();

    // ---- Phase C: greedy token (argmax of logits; softmax is monotone) ----
    if (tid < 16) {
      const float* row = lg + tid * Cc;
      float best = row[0]; int bi = 0;
      for (int c = 1; c < Cc; ++c) { float v = row[c]; if (v > best) { best = v; bi = c; } }
      tok[tid] = bi;
      tok_out[(size_t)(b0 + tid) * Tt + t] = bi;
    }
    __syncthreads();   // tok ready for next step's G gather
  }
}

// ---------------- host launch ----------------

extern "C" void kernel_launch(void* const* d_in, const int* in_sizes, int n_in,
                              void* d_out, int out_size, void* d_ws, size_t ws_size,
                              hipStream_t stream) {
  const float* feat   = (const float*)d_in[0];
  const float* W_ih   = (const float*)d_in[1];
  const float* W_hh   = (const float*)d_in[2];
  const float* b_ih   = (const float*)d_in[3];
  const float* b_hh   = (const float*)d_in[4];
  const float* W_proj = (const float*)d_in[5];
  const float* b_proj = (const float*)d_in[6];
  const float* embed  = (const float*)d_in[7];

  // workspace layout: Whh_bf16 | Wproj_bf16 | G  (~2.4 MB total)
  bfraw* Whh_bf   = (bfraw*)d_ws;
  bfraw* Wproj_bf = Whh_bf + (size_t)(3 * Dd) * Dd;
  float* G        = (float*)(Wproj_bf + (size_t)Cc * Dd);

  {
    int n = 3 * Dd * Dd;
    cvt_bf16_kernel<<<(n + 255) / 256, 256, 0, stream>>>(W_hh, Whh_bf, n);
  }
  {
    int n = Cc * Dd;
    cvt_bf16_kernel<<<(n + 255) / 256, 256, 0, stream>>>(W_proj, Wproj_bf, n);
  }
  {
    int n = 129 * 3 * Dd;
    build_G_kernel<<<(n + 255) / 256, 256, 0, stream>>>(embed, W_ih, b_ih, G);
  }

  float* logits_out = (float*)d_out;                                  // [B][C][T] f32
  int*   tok_out    = (int*)(logits_out + (size_t)Bsz * Cc * Tt);     // [B][T] i32

  gru_decode_kernel<<<Bsz / 16, 256, 0, stream>>>(
      feat, b_hh, Whh_bf, Wproj_bf, G, b_proj, logits_out, tok_out);
}
